// RAFTPose_37495064494650
// MI455X (gfx1250) — matrix operand order repaired and checked by reading, not compiled, not run on hardware
//
#include <hip/hip_runtime.h>
#include <hip/hip_bf16.h>
#include <cstdint>
#include <math.h>

typedef _Float16 h16;
typedef __attribute__((ext_vector_type(16))) _Float16 v16h;
typedef __attribute__((ext_vector_type(8)))  _Float16 v8h;
typedef __attribute__((ext_vector_type(8)))  float    v8f;

static constexpr int BB   = 2;
static constexpr int IH   = 256, IW = 640;
static constexpr int FH   = 32,  FW = 80;
static constexpr int P    = FH * FW;        // 2560
static constexpr int NS   = 16;
static constexpr int CCORR= 324;            // 81 * 4 levels
static constexpr int TIT  = 12;
static constexpr float PSTD = 0.01f;

static inline int ceil32(int x) { return (x + 31) & ~31; }
static inline int ceil64(int x) { return (x + 63) & ~63; }

__device__ __forceinline__ v16h cat16(v8h lo, v8h hi) {
  return __builtin_shufflevector(lo, hi, 0,1,2,3,4,5,6,7,8,9,10,11,12,13,14,15);
}
__device__ __forceinline__ v8h ldv8(const h16* p) { return *(const v8h*)p; }

// ---------------------------------------------------------------------------
// WMMA GEMM: C[M x N](f32) = A[Mpad x Kpad](f16 row-major) * B(f16, K16-interleaved)
// B layout: B[(k>>4)*N*16 + p*16 + (k&15)]  -> per-lane fragment = 2 aligned b128.
// 256 threads = 8 waves (2x4); 32x32 per wave -> block tile 64(M) x 128(N).
// op: 0 = bias, 1 = bias+relu, 2 = bias+sigmoid, 3 = bias+tanh
// ---------------------------------------------------------------------------
__global__ void __launch_bounds__(256) wmma_gemm(
    const h16* __restrict__ A, const h16* __restrict__ Bm,
    float* __restrict__ C, const float* __restrict__ bias,
    int M, int N, int Kpad, int lda, int ldc,
    long As, long Bs, long Cs, int op)
{
  int b = blockIdx.z;
  const h16* Ab = A + (size_t)b * As;
  const h16* Bb = Bm + (size_t)b * Bs;
  float*     Cb = C + (size_t)b * Cs;

  int w    = threadIdx.x >> 5;
  int lane = threadIdx.x & 31;
  int wm = w >> 2, wn = w & 3;
  int m0 = blockIdx.y * 64 + wm * 32;
  int n0 = blockIdx.x * 128 + wn * 32;
  if (n0 >= N) return;

  int lrow = lane & 15;
  int g    = lane >> 4;

  v8f acc00 = {}, acc01 = {}, acc10 = {}, acc11 = {};

  // A fragment: lane holds row; halves 0..15 -> K=k0+16g..k0+16g+15
  const h16* a0p = Ab + (size_t)(m0 + lrow) * lda + 16 * g;
  const h16* a1p = a0p + (size_t)16 * lda;
  // B fragment: lane holds col n0(+16)+lrow; 16 contiguous halves at kb=(k0>>4)+g
  const h16* bbase = Bb + (size_t)(n0 + lrow) * 16 + (size_t)g * N * 16;

  for (int k0 = 0; k0 < Kpad; k0 += 32) {
    v16h a0 = cat16(ldv8(a0p + k0), ldv8(a0p + k0 + 8));
    v16h a1 = cat16(ldv8(a1p + k0), ldv8(a1p + k0 + 8));
    const h16* bp = bbase + (size_t)(k0 >> 4) * N * 16;
    v16h b0 = cat16(ldv8(bp),       ldv8(bp + 8));
    v16h b1 = cat16(ldv8(bp + 256), ldv8(bp + 264));   // +16 columns
    acc00 = __builtin_amdgcn_wmma_f32_16x16x32_f16(false, a0, false, b0, (short)0, acc00, false, false);
    acc01 = __builtin_amdgcn_wmma_f32_16x16x32_f16(false, a0, false, b1, (short)0, acc01, false, false);
    acc10 = __builtin_amdgcn_wmma_f32_16x16x32_f16(false, a1, false, b0, (short)0, acc10, false, false);
    acc11 = __builtin_amdgcn_wmma_f32_16x16x32_f16(false, a1, false, b1, (short)0, acc11, false, false);
  }

  // C/D layout: VGPR v -> row m0+v (+8 for lanes 16..31), col = lane&15
  auto store = [&](v8f acc, int mrow, int ncol) {
    int colx = n0 + ncol + lrow;
#pragma unroll
    for (int v = 0; v < 8; v++) {
      int m = m0 + mrow + v + 8 * g;
      if (m < M) {
        float r = acc[v] + (bias ? bias[m] : 0.f);
        if (op == 1)      r = fmaxf(r, 0.f);
        else if (op == 2) r = 1.f / (1.f + expf(-r));
        else if (op == 3) r = tanhf(r);
        Cb[(size_t)m * ldc + colx] = r;
      }
    }
  };
  store(acc00, 0, 0);  store(acc01, 0, 16);
  store(acc10, 16, 0); store(acc11, 16, 16);
}

// ---------------------------------------------------------------------------
// im2col (f32 -> f16) into K16-interleaved layout:
// dst[b][kb][p][ki] with k = kb*16+ki; writes fully coalesced.
// ---------------------------------------------------------------------------
__global__ void im2col_f16(const float* __restrict__ src, h16* __restrict__ dst,
                           int C, int Hin, int Win, int Hout, int Wout,
                           int KH, int KW, int stride, int pad,
                           int Kpad, long sBS, long dBS)
{
  int b = blockIdx.z;
  long N = (long)Hout * Wout;
  long idx = blockIdx.x * (long)blockDim.x + threadIdx.x;
  if (idx >= (long)Kpad * N) return;
  int  ki = (int)(idx & 15);
  long t  = idx >> 4;
  long p  = t % N;
  int  kb = (int)(t / N);
  int  k  = kb * 16 + ki;
  float val = 0.f;
  int Kin = C * KH * KW;
  if (k < Kin) {
    int ci = k / (KH * KW);
    int rem = k % (KH * KW);
    int r = rem / KW, s = rem % KW;
    int py = (int)(p / Wout), px = (int)(p % Wout);
    int y = py * stride + r - pad;
    int x = px * stride + s - pad;
    if (y >= 0 && y < Hin && x >= 0 && x < Win)
      val = src[(size_t)b * sBS + ((size_t)ci * Hin + y) * Win + x];
  }
  dst[(size_t)b * dBS + idx] = (h16)val;
}

__global__ void pack_w(const float* __restrict__ w, h16* __restrict__ wp,
                       int M, int K, int Mp, int Kp)
{
  long idx = blockIdx.x * (long)blockDim.x + threadIdx.x;
  if (idx >= (long)Mp * Kp) return;
  int m = (int)(idx / Kp), k = (int)(idx % Kp);
  float v = (m < M && k < K) ? w[(size_t)m * K + k] : 0.f;
  wp[idx] = (h16)v;
}

// fmap_d_al (B,C=256,P) -> AT (B,P,256) f16 (row-major A) with 1/16 scale folded
__global__ void pack_at(const float* __restrict__ src, h16* __restrict__ dst)
{
  long idx = blockIdx.x * (long)blockDim.x + threadIdx.x;
  long tot = (long)BB * P * 256;
  if (idx >= tot) return;
  int b = (int)(idx / ((long)P * 256));
  long r = idx % ((long)P * 256);
  int p = (int)(r / 256), c = (int)(r % 256);
  dst[idx] = (h16)(src[((size_t)b * 256 + c) * P + p] * 0.0625f);
}

// ---------------------------------------------------------------------------
// instance norm (population var) + relu, in place. grid = (C, B), 256 thr.
// ---------------------------------------------------------------------------
__global__ void __launch_bounds__(256) instnorm_relu(float* __restrict__ x, int HW)
{
  int c = blockIdx.x, b = blockIdx.y, C = gridDim.x;
  float* t = x + ((size_t)b * C + c) * HW;
  float s = 0.f, s2 = 0.f;
  for (int i = threadIdx.x; i < HW; i += 256) { float v = t[i]; s += v; s2 += v * v; }
  __shared__ float r1[256], r2[256];
  r1[threadIdx.x] = s; r2[threadIdx.x] = s2; __syncthreads();
  for (int st = 128; st > 0; st >>= 1) {
    if (threadIdx.x < st) { r1[threadIdx.x] += r1[threadIdx.x + st];
                            r2[threadIdx.x] += r2[threadIdx.x + st]; }
    __syncthreads();
  }
  float mean = r1[0] / HW;
  float var  = r2[0] / HW - mean * mean;
  float inv  = rsqrtf(var + 1e-5f);
  for (int i = threadIdx.x; i < HW; i += 256)
    t[i] = fmaxf((t[i] - mean) * inv, 0.f);
}

__global__ void avgpool2_k(const float* __restrict__ src, float* __restrict__ dst,
                           int Hl, int Wl)
{
  int H2 = Hl >> 1, W2 = Wl >> 1;
  long idx = blockIdx.x * (long)blockDim.x + threadIdx.x;
  long tot = (long)BB * P * H2 * W2;
  if (idx >= tot) return;
  long bp = idx / ((long)H2 * W2);
  int  r  = (int)(idx % ((long)H2 * W2));
  int oy = r / W2, ox = r % W2;
  const float* s = src + bp * ((size_t)Hl * Wl);
  float v = 0.25f * (s[(2*oy) * Wl + 2*ox]     + s[(2*oy) * Wl + 2*ox + 1] +
                     s[(2*oy+1) * Wl + 2*ox]   + s[(2*oy+1) * Wl + 2*ox + 1]);
  dst[idx] = v;
}

// ---------------------------------------------------------------------------
// geometry / sampling / projection
// ---------------------------------------------------------------------------
__global__ void compute_pts(const float* __restrict__ depth,
                            const float* __restrict__ Kd, float* __restrict__ pts)
{
  int idx = blockIdx.x * blockDim.x + threadIdx.x;
  if (idx >= BB * P) return;
  int b = idx / P, p = idx % P;
  int v = p / FW, u = p % FW;
  float fxs = Kd[b*9+0] * 0.125f, cxs = Kd[b*9+2] * 0.125f;
  float fys = Kd[b*9+4] * 0.125f, cys = Kd[b*9+5] * 0.125f;
  float d = depth[(size_t)b * IH * IW + (size_t)(v * 8) * IW + (u * 8)];
  pts[((size_t)b*3 + 0) * P + p] = d * ((float)u - cxs) / fxs;
  pts[((size_t)b*3 + 1) * P + p] = d * ((float)v - cys) / fys;
  pts[((size_t)b*3 + 2) * P + p] = d;
}

__device__ __forceinline__ uint32_t hashu(uint32_t x) {
  x ^= x >> 16; x *= 0x7feb352du; x ^= x >> 15; x *= 0x846ca68bu; x ^= x >> 16;
  return x;
}
__device__ __forceinline__ float urand(uint32_t s) {
  return (float)(hashu(s) >> 8) * (1.f / 16777216.f);
}
__device__ __forceinline__ float nrand(uint32_t s) {
  float u1 = fmaxf(urand(2u * s + 1u), 1e-7f);
  float u2 = urand(2u * s + 2u);
  return sqrtf(-2.f * logf(u1)) * cosf(6.2831853f * u2);
}

__global__ void sample_poses(const float* __restrict__ pose,
                             float* __restrict__ Rm, float* __restrict__ Ts, int iter)
{
  int i = blockIdx.x * blockDim.x + threadIdx.x;
  if (i >= BB * NS) return;
  int b = i / NS, n = i % NS;
  uint32_t base = (uint32_t)(((iter * BB + b) * NS + n) * 16 + 7);
  float rn[3], tn[3];
  for (int k = 0; k < 3; k++) {
    rn[k] = (n == 0) ? 0.f : nrand(base + k);
    tn[k] = (n == 0) ? 0.f : nrand(base + 8 + k);
  }
  float pw = pose[b*7+0], px = pose[b*7+1], py = pose[b*7+2], pz = pose[b*7+3];
  float d0 = 1.f, d1 = rn[0]*PSTD, d2 = rn[1]*PSTD, d3 = rn[2]*PSTD;
  float dn = rsqrtf(d0*d0 + d1*d1 + d2*d2 + d3*d3);
  d0 *= dn; d1 *= dn; d2 *= dn; d3 *= dn;
  float w = pw*d0 - px*d1 - py*d2 - pz*d3;
  float x = pw*d1 + px*d0 + py*d3 - pz*d2;
  float y = pw*d2 - px*d3 + py*d0 + pz*d1;
  float z = pw*d3 + px*d2 - py*d1 + pz*d0;
  float qn = rsqrtf(w*w + x*x + y*y + z*z);
  w *= qn; x *= qn; y *= qn; z *= qn;
  float* R = Rm + (size_t)i * 9;
  R[0] = 1.f - 2.f*(y*y + z*z); R[1] = 2.f*(x*y - w*z);       R[2] = 2.f*(x*z + w*y);
  R[3] = 2.f*(x*y + w*z);       R[4] = 1.f - 2.f*(x*x + z*z); R[5] = 2.f*(y*z - w*x);
  R[6] = 2.f*(x*z - w*y);       R[7] = 2.f*(y*z + w*x);       R[8] = 1.f - 2.f*(x*x + y*y);
  for (int k = 0; k < 3; k++) Ts[(size_t)i*3 + k] = pose[b*7 + 4 + k] + tn[k]*PSTD;
}

__global__ void project_k(const float* __restrict__ Rm, const float* __restrict__ Ts,
                          const float* __restrict__ pts, const float* __restrict__ Krgb,
                          float* __restrict__ coords)
{
  long idx = blockIdx.x * (long)blockDim.x + threadIdx.x;
  if (idx >= (long)BB * NS * P) return;
  int b = (int)(idx / ((long)NS * P));
  int r = (int)(idx % ((long)NS * P));
  int n = r / P, p = r % P;
  const float* R = Rm + ((size_t)b * NS + n) * 9;
  const float* t = Ts + ((size_t)b * NS + n) * 3;
  float px = pts[((size_t)b*3+0)*P + p];
  float py = pts[((size_t)b*3+1)*P + p];
  float pz = pts[((size_t)b*3+2)*P + p];
  float X = R[0]*px + R[1]*py + R[2]*pz + t[0];
  float Y = R[3]*px + R[4]*py + R[5]*pz + t[1];
  float Z = R[6]*px + R[7]*py + R[8]*pz + t[2];
  float fxs = Krgb[b*9+0]*0.125f, cxs = Krgb[b*9+2]*0.125f;
  float fys = Krgb[b*9+4]*0.125f, cys = Krgb[b*9+5]*0.125f;
  float u_ = fxs*X + cxs*Z, v_ = fys*Y + cys*Z;
  float cx = -10000.f, cy = -10000.f;
  if (Z > 0.001f) { float zi = 1.f / fmaxf(Z, 0.001f); cx = u_*zi; cy = v_*zi; }
  coords[(((size_t)b*NS + n)*P + p)*2 + 0] = cx;
  coords[(((size_t)b*NS + n)*P + p)*2 + 1] = cy;
}

__device__ __forceinline__ float bilin(const float* __restrict__ t, int Hl, int Wl,
                                       float x, float y)
{
  float x0f = floorf(x), y0f = floorf(y);
  float fx = x - x0f, fy = y - y0f;
  int x0 = (int)x0f, y0 = (int)y0f, x1 = x0 + 1, y1 = y0 + 1;
  bool vx0 = (x0 >= 0) && (x0 < Wl), vx1 = (x1 >= 0) && (x1 < Wl);
  bool vy0 = (y0 >= 0) && (y0 < Hl), vy1 = (y1 >= 0) && (y1 < Hl);
  float v00 = (vx0 && vy0) ? t[y0 * Wl + x0] : 0.f;
  float v10 = (vx1 && vy0) ? t[y0 * Wl + x1] : 0.f;
  float v01 = (vx0 && vy1) ? t[y1 * Wl + x0] : 0.f;
  float v11 = (vx1 && vy1) ? t[y1 * Wl + x1] : 0.f;
  return v00*(1.f-fx)*(1.f-fy) + v10*fx*(1.f-fy) + v01*(1.f-fx)*fy + v11*fx*fy;
}

// confidence[b,n] = mean over (p, 324 features) of pyramid lookups
__global__ void __launch_bounds__(256) confidence_k(
    const float* __restrict__ c0, const float* __restrict__ c1,
    const float* __restrict__ c2, const float* __restrict__ c3,
    const float* __restrict__ coords, float* __restrict__ conf)
{
  int n = blockIdx.x, b = blockIdx.y;
  const float* lv[4] = {c0, c1, c2, c3};
  float s = 0.f;
  for (int p = threadIdx.x; p < P; p += 256) {
    float cx = coords[(((size_t)b*NS + n)*P + p)*2 + 0];
    float cy = coords[(((size_t)b*NS + n)*P + p)*2 + 1];
    for (int l = 0; l < 4; l++) {
      int Hl = FH >> l, Wl = FW >> l;
      float sc = 1.f / (float)(1 << l);
      const float* base = lv[l] + ((size_t)b * P + p) * (size_t)(Hl * Wl);
      float px = cx * sc, py = cy * sc;
      for (int dy = -4; dy <= 4; dy++)
        for (int dx = -4; dx <= 4; dx++)
          s += bilin(base, Hl, Wl, px + (float)dx, py + (float)dy);
    }
  }
  __shared__ float red[256];
  red[threadIdx.x] = s; __syncthreads();
  for (int st = 128; st > 0; st >>= 1) {
    if (threadIdx.x < st) red[threadIdx.x] += red[threadIdx.x + st];
    __syncthreads();
  }
  if (threadIdx.x == 0) conf[b * NS + n] = red[0] / ((float)P * (float)CCORR);
}

__global__ void argmax_k(const float* __restrict__ conf, int* __restrict__ best)
{
  int b = threadIdx.x;
  if (b >= BB) return;
  float mx = conf[b * NS]; int bi = 0;
  for (int n = 1; n < NS; n++) {
    float v = conf[b * NS + n];
    if (v > mx) { mx = v; bi = n; }
  }
  best[b] = bi;
}

// best_corr (B, 324, P): channel = l*81 + (dy+4)*9 + (dx+4)
__global__ void gather_best(const float* __restrict__ c0, const float* __restrict__ c1,
                            const float* __restrict__ c2, const float* __restrict__ c3,
                            const float* __restrict__ coords, const int* __restrict__ best,
                            float* __restrict__ bc)
{
  long idx = blockIdx.x * (long)blockDim.x + threadIdx.x;
  long tot = (long)BB * CCORR * P;
  if (idx >= tot) return;
  int b = (int)(idx / ((long)CCORR * P));
  int r = (int)(idx % ((long)CCORR * P));
  int c = r / P, p = r % P;
  int n = best[b];
  int l = c / 81, d = c % 81;
  int dy = d / 9 - 4, dx = d % 9 - 4;
  const float* lv = (l == 0) ? c0 : (l == 1) ? c1 : (l == 2) ? c2 : c3;
  int Hl = FH >> l, Wl = FW >> l;
  float sc = 1.f / (float)(1 << l);
  float cx = coords[(((size_t)b*NS + n)*P + p)*2 + 0] * sc + (float)dx;
  float cy = coords[(((size_t)b*NS + n)*P + p)*2 + 1] * sc + (float)dy;
  const float* base = lv + ((size_t)b * P + p) * (size_t)(Hl * Wl);
  bc[((size_t)b * CCORR + c) * P + p] = bilin(base, Hl, Wl, cx, cy);
}

// ---------------------------------------------------------------------------
// elementwise / reductions for the GRU
// ---------------------------------------------------------------------------
__global__ void copy_chan(const float* __restrict__ src, float* __restrict__ dst,
                          long chunk, long sstride, long dstride)
{
  long idx = blockIdx.x * (long)blockDim.x + threadIdx.x;
  if (idx >= (long)BB * chunk) return;
  long b = idx / chunk, r = idx % chunk;
  dst[b * dstride + r] = src[b * sstride + r];
}

__global__ void mul_into(const float* __restrict__ a, const float* __restrict__ b,
                         float* __restrict__ dst, long chunk, long dstride)
{
  long idx = blockIdx.x * (long)blockDim.x + threadIdx.x;
  if (idx >= (long)BB * chunk) return;
  long bb = idx / chunk, r = idx % chunk;
  dst[bb * dstride + r] = a[bb * chunk + r] * b[bb * chunk + r];
}

__global__ void gru_update(float* __restrict__ h, const float* __restrict__ z,
                           const float* __restrict__ q, long tot)
{
  long idx = blockIdx.x * (long)blockDim.x + threadIdx.x;
  if (idx >= tot) return;
  float zz = z[idx];
  h[idx] = (1.f - zz) * h[idx] + zz * q[idx];
}

__global__ void __launch_bounds__(256) spatial_mean(const float* __restrict__ x,
                                                    float* __restrict__ out, int HW)
{
  int c = blockIdx.x, b = blockIdx.y, C = gridDim.x;
  const float* t = x + ((size_t)b * C + c) * HW;
  float s = 0.f;
  for (int i = threadIdx.x; i < HW; i += 256) s += t[i];
  __shared__ float red[256];
  red[threadIdx.x] = s; __syncthreads();
  for (int st = 128; st > 0; st >>= 1) {
    if (threadIdx.x < st) red[threadIdx.x] += red[threadIdx.x + st];
    __syncthreads();
  }
  if (threadIdx.x == 0) out[b * C + c] = red[0] / (float)HW;
}

__global__ void pose_update(float* __restrict__ pose, const float* __restrict__ dpose)
{
  int b = threadIdx.x;
  if (b >= BB) return;
  float q0 = pose[b*7+0], q1 = pose[b*7+1], q2 = pose[b*7+2], q3 = pose[b*7+3];
  float d0 = dpose[b*7+0] + 1.f, d1 = dpose[b*7+1], d2 = dpose[b*7+2], d3 = dpose[b*7+3];
  float w = q0*d0 - q1*d1 - q2*d2 - q3*d3;
  float x = q0*d1 + q1*d0 + q2*d3 - q3*d2;
  float y = q0*d2 - q1*d3 + q2*d0 + q3*d1;
  float z = q0*d3 + q1*d2 - q2*d1 + q3*d0;
  float inv = rsqrtf(w*w + x*x + y*y + z*z);
  pose[b*7+0] = w*inv; pose[b*7+1] = x*inv; pose[b*7+2] = y*inv; pose[b*7+3] = z*inv;
  pose[b*7+4] += dpose[b*7+4];
  pose[b*7+5] += dpose[b*7+5];
  pose[b*7+6] += dpose[b*7+6];
}

__global__ void fill_zero(float* __restrict__ p, long n)
{
  long i = blockIdx.x * (long)blockDim.x + threadIdx.x;
  if (i < n) p[i] = 0.f;
}

__global__ void copy_f32(float* __restrict__ dst, const float* __restrict__ src, long n)
{
  long i = blockIdx.x * (long)blockDim.x + threadIdx.x;
  if (i < n) dst[i] = src[i];
}

// ---------------------------------------------------------------------------
// host driver
// ---------------------------------------------------------------------------
extern "C" void kernel_launch(void* const* d_in, const int* in_sizes, int n_in,
                              void* d_out, int out_size, void* d_ws, size_t ws_size,
                              hipStream_t stream)
{
  (void)in_sizes; (void)n_in; (void)out_size; (void)ws_size;

  const float* image  = (const float*)d_in[0];
  const float* depth  = (const float*)d_in[1];
  const float* Krgb   = (const float*)d_in[2];
  const float* Kdep   = (const float*)d_in[3];
  const float* ipose  = (const float*)d_in[4];
  auto PW = [&](int i) { return (const float*)d_in[5 + i]; };

  char* wsb = (char*)d_ws;
  size_t off = 0;
  auto alloc = [&](size_t bytes) -> void* {
    void* r = wsb + off;
    off = (off + bytes + 255) & ~(size_t)255;
    return r;
  };

  const int N1 = 128 * 320;   // 40960
  const int N2 = 64 * 160;    // 10240

  float* corr0 = (float*)alloc((size_t)BB * P * P * 4);
  float* corr1 = (float*)alloc((size_t)BB * P * (P / 4) * 4);
  float* corr2 = (float*)alloc((size_t)BB * P * (P / 16) * 4);
  float* corr3 = (float*)alloc((size_t)BB * P * (P / 64) * 4);
  float* context = (float*)alloc((size_t)BB * 64 * P * 4);
  float* hidden  = (float*)alloc((size_t)BB * 128 * P * 4);
  float* pts     = (float*)alloc((size_t)BB * 3 * P * 4);
  float* pose    = (float*)alloc((size_t)BB * 7 * 4);
  float* Rmb     = (float*)alloc((size_t)BB * NS * 9 * 4);
  float* Tsb     = (float*)alloc((size_t)BB * NS * 3 * 4);
  float* confb   = (float*)alloc((size_t)BB * NS * 4);
  int*   bestb   = (int*)  alloc((size_t)BB * 4);
  float* coords  = (float*)alloc((size_t)BB * NS * P * 2 * 4);
  float* bcorr   = (float*)alloc((size_t)BB * CCORR * P * 4);
  float* hx      = (float*)alloc((size_t)BB * 320 * P * 4);
  float* m1      = (float*)alloc((size_t)BB * 128 * P * 4);
  float* m2      = (float*)alloc((size_t)BB * 128 * P * 4);
  float* zg      = (float*)alloc((size_t)BB * 128 * P * 4);
  float* rg      = (float*)alloc((size_t)BB * 128 * P * 4);
  float* qg      = (float*)alloc((size_t)BB * 128 * P * 4);
  float* outmap  = (float*)alloc((size_t)BB * 7 * P * 4);
  float* dpose   = (float*)alloc((size_t)BB * 7 * 4);
  float* f1  = (float*)alloc((size_t)BB * 64 * N1 * 4);
  float* f2  = (float*)alloc((size_t)BB * 128 * N2 * 4);
  float* f3  = (float*)alloc((size_t)BB * 256 * P * 4);
  float* frgb= (float*)alloc((size_t)BB * 256 * P * 4);
  float* d1  = (float*)alloc((size_t)BB * 32 * N1 * 4);
  float* d2  = (float*)alloc((size_t)BB * 32 * N2 * 4);
  float* fd  = (float*)alloc((size_t)BB * 32 * P * 4);
  float* fal = (float*)alloc((size_t)BB * 256 * P * 4);
  h16* AT16  = (h16*)alloc((size_t)BB * P * 256 * 2);
  h16* B16   = (h16*)alloc((size_t)BB * 256 * P * 2);
  h16* Xcol  = (h16*)alloc((size_t)BB * 2880 * 2560 * 2);

  // packed weights (M padded to 64 for the 64-row block tile)
  auto walloc = [&](int M, int K) { return (h16*)alloc((size_t)ceil64(M) * ceil32(K) * 2); };
  h16 *Wiw1 = walloc(64, 147),  *Wiw2 = walloc(128, 576), *Wiw3 = walloc(256, 1152);
  h16 *Wiw4 = walloc(256, 256), *Wdw1 = walloc(32, 49),   *Wdw2 = walloc(32, 288);
  h16 *Wdw3 = walloc(32, 288),  *Waw  = walloc(256, 32),  *Wcw  = walloc(64, 32);
  h16 *Wpc1 = walloc(128, 324), *Wpc2 = walloc(128, 1152),*Wpc3 = walloc(128, 1152);
  h16 *Wgz  = walloc(128, 2880),*Wgr  = walloc(128, 2880),*Wgq  = walloc(128, 2880);
  h16 *Whw1 = walloc(128, 1152),*Whw2 = walloc(7, 128);

  auto packw = [&](const float* w, h16* wp, int M, int K) {
    int Mp = ceil64(M), Kp = ceil32(K);
    long tot = (long)Mp * Kp;
    pack_w<<<dim3((unsigned)((tot + 255) / 256)), 256, 0, stream>>>(w, wp, M, K, Mp, Kp);
  };
  packw(PW(0),  Wiw1, 64, 147);   packw(PW(2),  Wiw2, 128, 576);
  packw(PW(4),  Wiw3, 256, 1152); packw(PW(6),  Wiw4, 256, 256);
  packw(PW(8),  Wdw1, 32, 49);    packw(PW(10), Wdw2, 32, 288);
  packw(PW(12), Wdw3, 32, 288);   packw(PW(14), Waw, 256, 32);
  packw(PW(16), Wcw, 64, 32);     packw(PW(18), Wpc1, 128, 324);
  packw(PW(20), Wpc2, 128, 1152); packw(PW(22), Wpc3, 128, 1152);
  packw(PW(24), Wgz, 128, 2880);  packw(PW(26), Wgr, 128, 2880);
  packw(PW(28), Wgq, 128, 2880);  packw(PW(30), Whw1, 128, 1152);
  packw(PW(32), Whw2, 7, 128);

  auto im2col = [&](const float* src, h16* dst, int C, int Hin, int Win,
                    int Hout, int Wout, int KH, int KW, int stride, int pad, int Kp) {
    long N = (long)Hout * Wout;
    long tot = (long)Kp * N;
    dim3 grid((unsigned)((tot + 255) / 256), 1, BB);
    im2col_f16<<<grid, 256, 0, stream>>>(src, dst, C, Hin, Win, Hout, Wout,
                                         KH, KW, stride, pad, Kp,
                                         (long)C * Hin * Win, tot);
  };
  auto gemm = [&](const h16* A, const h16* Bm, float* C, const float* bias,
                  int M, int N, int Kp, int lda, int ldc,
                  long As, long Bs, long Cs, int op) {
    dim3 grid((unsigned)(N / 128), (unsigned)((M + 63) / 64), BB);
    wmma_gemm<<<grid, 256, 0, stream>>>(A, Bm, C, bias, M, N, Kp,
                                        lda, ldc, As, Bs, Cs, op);
  };
  auto conv = [&](const float* src, float* out, const h16* Wp, const float* bias,
                  int C, int Hin, int Win, int Hout, int Wout,
                  int KH, int KW, int stride, int pad, int M, int op) {
    int Kp = ceil32(C * KH * KW);
    im2col(src, Xcol, C, Hin, Win, Hout, Wout, KH, KW, stride, pad, Kp);
    int N = Hout * Wout;
    gemm(Wp, Xcol, out, bias, M, N, Kp, Kp, N, 0, (long)Kp * N, (long)M * N, op);
  };
  auto inorm = [&](float* x, int C, int HW) {
    instnorm_relu<<<dim3(C, BB), 256, 0, stream>>>(x, HW);
  };

  // ---------------- encoder ----------------
  conv(image, f1, Wiw1, PW(1), 3, 256, 640, 128, 320, 7, 7, 2, 2, 64, 0);
  inorm(f1, 64, N1);
  conv(f1, f2, Wiw2, PW(3), 64, 128, 320, 64, 160, 3, 3, 2, 0, 128, 0);
  inorm(f2, 128, N2);
  conv(f2, f3, Wiw3, PW(5), 128, 64, 160, 32, 80, 3, 3, 2, 0, 256, 0);
  inorm(f3, 256, P);
  conv(f3, frgb, Wiw4, PW(7), 256, 32, 80, 32, 80, 1, 1, 1, 0, 256, 0);

  conv(depth, d1, Wdw1, PW(9), 1, 256, 640, 128, 320, 7, 7, 2, 2, 32, 0);
  inorm(d1, 32, N1);
  conv(d1, d2, Wdw2, PW(11), 32, 128, 320, 64, 160, 3, 3, 2, 0, 32, 0);
  inorm(d2, 32, N2);
  conv(d2, fd, Wdw3, PW(13), 32, 64, 160, 32, 80, 3, 3, 2, 0, 32, 0);
  inorm(fd, 32, P);

  {   // aw + cw share the same 1x1 im2col of fmap_depth
    int Kp = 32;
    im2col(fd, Xcol, 32, 32, 80, 32, 80, 1, 1, 1, 0, Kp);
    gemm(Waw, Xcol, fal, PW(15), 256, P, Kp, Kp, P, 0, (long)Kp * P, (long)256 * P, 0);
    gemm(Wcw, Xcol, context, PW(17), 64, P, Kp, Kp, P, 0, (long)Kp * P, (long)64 * P, 1);
    inorm(fal, 256, P);
  }

  // ---------------- correlation volume + pyramid ----------------
  {
    long tot = (long)BB * P * 256;
    pack_at<<<dim3((unsigned)((tot + 255) / 256)), 256, 0, stream>>>(fal, AT16);
    im2col(frgb, B16, 256, 32, 80, 32, 80, 1, 1, 1, 0, 256);   // f32->f16 cast, K16 layout
    gemm(AT16, B16, corr0, nullptr, P, P, 256, 256, P,
         (long)P * 256, (long)256 * P, (long)P * P, 0);
  }
  {
    long t1 = (long)BB * P * (P / 4);
    avgpool2_k<<<dim3((unsigned)((t1 + 255) / 256)), 256, 0, stream>>>(corr0, corr1, 32, 80);
    long t2 = (long)BB * P * (P / 16);
    avgpool2_k<<<dim3((unsigned)((t2 + 255) / 256)), 256, 0, stream>>>(corr1, corr2, 16, 40);
    long t3 = (long)BB * P * (P / 64);
    avgpool2_k<<<dim3((unsigned)((t3 + 255) / 256)), 256, 0, stream>>>(corr2, corr3, 8, 20);
  }

  // ---------------- geometry + state init ----------------
  compute_pts<<<dim3((BB * P + 255) / 256), 256, 0, stream>>>(depth, Kdep, pts);
  copy_f32<<<1, 64, 0, stream>>>(pose, ipose, BB * 7);
  {
    long n = (long)BB * 128 * P;
    fill_zero<<<dim3((unsigned)((n + 255) / 256)), 256, 0, stream>>>(hidden, n);
  }

  // ---------------- GRU iterations ----------------
  const long chunk128 = (long)128 * P, s320 = (long)320 * P;
  for (int it = 0; it < TIT; it++) {
    sample_poses<<<1, BB * NS, 0, stream>>>(pose, Rmb, Tsb, it);
    {
      long n = (long)BB * NS * P;
      project_k<<<dim3((unsigned)((n + 255) / 256)), 256, 0, stream>>>(
          Rmb, Tsb, pts, Krgb, coords);
    }
    confidence_k<<<dim3(NS, BB), 256, 0, stream>>>(corr0, corr1, corr2, corr3, coords, confb);
    argmax_k<<<1, 32, 0, stream>>>(confb, bestb);
    {
      long n = (long)BB * CCORR * P;
      gather_best<<<dim3((unsigned)((n + 255) / 256)), 256, 0, stream>>>(
          corr0, corr1, corr2, corr3, coords, bestb, bcorr);
    }
    // motion encoder
    conv(bcorr, m1, Wpc1, PW(19), CCORR, 32, 80, 32, 80, 1, 1, 1, 0, 128, 1);
    conv(m1, m2, Wpc2, PW(21), 128, 32, 80, 32, 80, 3, 3, 1, 1, 128, 1);
    conv(m2, m1, Wpc3, PW(23), 128, 32, 80, 32, 80, 3, 3, 1, 1, 128, 1);
    // hx = [hidden | m | context]
    {
      unsigned g = (unsigned)(((long)BB * chunk128 + 255) / 256);
      copy_chan<<<g, 256, 0, stream>>>(hidden, hx, chunk128, chunk128, s320);
      copy_chan<<<g, 256, 0, stream>>>(m1, hx + (size_t)128 * P, chunk128, chunk128, s320);
      unsigned g2 = (unsigned)(((long)BB * 64 * P + 255) / 256);
      copy_chan<<<g2, 256, 0, stream>>>(context, hx + (size_t)256 * P,
                                        (long)64 * P, (long)64 * P, s320);
    }
    // gates: one im2col feeds gz and gr
    {
      int Kp = 2880;
      im2col(hx, Xcol, 320, 32, 80, 32, 80, 3, 3, 1, 1, Kp);
      gemm(Wgz, Xcol, zg, PW(25), 128, P, Kp, Kp, P, 0, (long)Kp * P, chunk128, 2);
      gemm(Wgr, Xcol, rg, PW(27), 128, P, Kp, Kp, P, 0, (long)Kp * P, chunk128, 2);
      unsigned g = (unsigned)(((long)BB * chunk128 + 255) / 256);
      mul_into<<<g, 256, 0, stream>>>(rg, hidden, hx, chunk128, s320);
      im2col(hx, Xcol, 320, 32, 80, 32, 80, 3, 3, 1, 1, Kp);
      gemm(Wgq, Xcol, qg, PW(29), 128, P, Kp, Kp, P, 0, (long)Kp * P, chunk128, 3);
    }
    {
      long n = (long)BB * chunk128;
      gru_update<<<dim3((unsigned)((n + 255) / 256)), 256, 0, stream>>>(hidden, zg, qg, n);
    }
    // pose head
    conv(hidden, m2, Whw1, PW(31), 128, 32, 80, 32, 80, 3, 3, 1, 1, 128, 1);
    conv(m2, outmap, Whw2, PW(33), 128, 32, 80, 32, 80, 1, 1, 1, 0, 7, 0);
    spatial_mean<<<dim3(7, BB), 256, 0, stream>>>(outmap, dpose, P);
    pose_update<<<1, 32, 0, stream>>>(pose, dpose);
  }

  copy_f32<<<1, 64, 0, stream>>>((float*)d_out, pose, BB * 7);
}
